// MixtureOfExperts_33011118637071
// MI455X (gfx1250) — matrix operand order repaired and checked by reading, not compiled, not run on hardware
//
#include <hip/hip_runtime.h>
#include <hip/hip_bf16.h>

typedef __attribute__((ext_vector_type(16))) _Float16 v16h;
typedef __attribute__((ext_vector_type(8)))  float    v8f;

#define NROWS 16384
#define DDIM  256
#define EEXP  8
#define HHID  128
#define LDSPITCH 264   // 256 + 8 f16 pad -> 528B row stride, breaks 256B bank period

#if defined(__has_builtin)
#if __has_builtin(__builtin_amdgcn_global_load_async_to_lds_b128)
#define HAVE_ASYNC_LDS 1
#endif
#endif
#ifndef HAVE_ASYNC_LDS
#define HAVE_ASYNC_LDS 0
#endif

#if HAVE_ASYNC_LDS
typedef int vi4 __attribute__((__vector_size__(4 * sizeof(int))));
typedef __attribute__((address_space(1))) vi4* as1_vi4p;
typedef __attribute__((address_space(3))) vi4* as3_vi4p;
#endif

static __device__ __forceinline__ v8f wmma16(v16h a, v16h b, v8f c) {
    return __builtin_amdgcn_wmma_f32_16x16x32_f16(false, a, false, b, (short)0, c, false, false);
}

// A fragment (16x32 f16, M=lane%16) from fp32 row-major source, with convert.
static __device__ __forceinline__ v16h loadAfragF32(const float* __restrict__ rowPtr, int t, int lhi) {
    const int k0 = t * 32 + lhi * 8;
    const int k1 = k0 + 16;
    v16h a;
#pragma unroll
    for (int j = 0; j < 8; ++j) {
        a[j]     = (_Float16)rowPtr[k0 + j];
        a[8 + j] = (_Float16)rowPtr[k1 + j];
    }
    return a;
}

// A fragment from f16 LDS tile hw[16][LDSPITCH] (row-major, M=lane%16).
static __device__ __forceinline__ v16h loadAfragLDS(const _Float16* hbase, int lr, int lhi, int t) {
    const int k0 = t * 32 + lhi * 8;
    union { float4 f[2]; v16h h; } u;
    u.f[0] = *(const float4*)(hbase + lr * LDSPITCH + k0);
    u.f[1] = *(const float4*)(hbase + lr * LDSPITCH + k0 + 16);
    return u.h;
}

// B fragment (32x16 f16, N=lane%16) from LDS block [16][LDSPITCH].
static __device__ __forceinline__ v16h loadBfragLDS(const _Float16* base, int lr, int lhi, int t) {
    union { float4 f[2]; v16h h; } u;
    const float4* p = (const float4*)(base + lr * LDSPITCH + t * 32 + lhi * 8);
    u.f[0] = p[0];
    u.f[1] = p[2];
    return u.h;
}

// B fragment directly from global f16 [N][K].
static __device__ __forceinline__ v16h loadBfragG(const _Float16* __restrict__ W, int rowStride,
                                                  int lr, int lhi, int t) {
    union { float4 f[2]; v16h h; } u;
    const float4* p = (const float4*)(W + lr * rowStride + t * 32 + lhi * 8);
    u.f[0] = p[0];
    u.f[1] = p[2];
    return u.h;
}

// ---- double-buffered cooperative B-tile staging (16 N x 256 K f16, 128 threads) ----
struct BStage { float4 v[4]; };

static __device__ __forceinline__ void tileFetch(const _Float16* __restrict__ src, _Float16* buf,
                                                 int tid, BStage& st) {
    const int n  = tid >> 3;          // 0..15
    const int kc = (tid & 7) * 32;    // 0..224
#if HAVE_ASYNC_LDS
    const _Float16* g = src + n * DDIM + kc;
    _Float16*       l = buf + n * LDSPITCH + kc;
    as1_vi4p gp = (as1_vi4p)(unsigned long long)(const void*)g;
    as3_vi4p lp = (as3_vi4p)(unsigned int)(unsigned long long)(void*)l;
    __builtin_amdgcn_global_load_async_to_lds_b128(gp, lp, 0, 0);
    __builtin_amdgcn_global_load_async_to_lds_b128(gp, lp, 16, 0);
    __builtin_amdgcn_global_load_async_to_lds_b128(gp, lp, 32, 0);
    __builtin_amdgcn_global_load_async_to_lds_b128(gp, lp, 48, 0);
    (void)st;
#else
    const float4* s = (const float4*)(src + n * DDIM + kc);
    st.v[0] = s[0]; st.v[1] = s[1]; st.v[2] = s[2]; st.v[3] = s[3];
    (void)buf;
#endif
}

static __device__ __forceinline__ void tileCommit(_Float16* buf, int tid, BStage& st) {
#if !HAVE_ASYNC_LDS
    const int n  = tid >> 3;
    const int kc = (tid & 7) * 32;
    float4* d = (float4*)(buf + n * LDSPITCH + kc);
    d[0] = st.v[0]; d[1] = st.v[1]; d[2] = st.v[2]; d[3] = st.v[3];
#else
    (void)buf; (void)tid; (void)st;
#endif
}

static __device__ __forceinline__ void asyncFence() {
#if HAVE_ASYNC_LDS
#if __has_builtin(__builtin_amdgcn_s_wait_asynccnt)
    __builtin_amdgcn_s_wait_asynccnt(0);
#else
    asm volatile("s_wait_asynccnt 0" ::: "memory");
#endif
#endif
}

// Transpose + fp32->f16 convert: dst[b][n][k] = (n < Nsrc) ? src[b][k][n] : 0
__global__ void moe_xpose_cvt(const float* __restrict__ src, _Float16* __restrict__ dst,
                              int K, int Nsrc, int Ndst, int total) {
    int i = blockIdx.x * blockDim.x + threadIdx.x;
    if (i >= total) return;
    int k = i % K;
    int n = (i / K) % Ndst;
    int b = i / (K * Ndst);
    dst[i] = (n < Nsrc) ? (_Float16)src[(b * K + k) * Nsrc + n] : (_Float16)0.0f;
}

__global__ __launch_bounds__(128) void moe_main_kernel(
    const float* __restrict__ x,
    const _Float16* __restrict__ Wt1,   // [E][256][256] f16, pre-transposed (N-major)
    const _Float16* __restrict__ Wt2,   // [E][256][256]
    const _Float16* __restrict__ Wg1t,  // [128][256]
    const _Float16* __restrict__ Wg2t,  // [16][128] (cols 8..15 zero)
    const float* __restrict__ bg1, const float* __restrict__ bg2,
    const float* __restrict__ b1,  const float* __restrict__ b2,
    float* __restrict__ out) {
    __shared__ __align__(16) _Float16 ldsB[2][16 * LDSPITCH];    // double-buffered B tile
    __shared__ __align__(16) _Float16 ldsH[4][16 * LDSPITCH];    // per-wave h tile
    __shared__ float ldsG[64][EEXP];                             // gate weights

    const int tid  = threadIdx.x;
    const int w    = tid >> 5;
    const int lane = tid & 31;
    const int lr   = lane & 15;
    const int lhi  = lane >> 4;
    const int rowBase = blockIdx.x * 64 + w * 16;

    // Preload this wave's 16 rows of x as A fragments (K = 256 -> 8 fragments)
    const float* xrow = x + (size_t)(rowBase + lr) * DDIM;
    v16h xf[8];
#pragma unroll
    for (int t = 0; t < 8; ++t) xf[t] = loadAfragF32(xrow, t, lhi);

    _Float16* hw = ldsH[w];

    // -------- gate GEMM1: hid = relu(x @ Wg1 + bg1), H = 128 --------
    {
        BStage st;
        tileFetch(Wg1t, ldsB[0], tid, st);
        tileCommit(ldsB[0], tid, st);
        asyncFence();
        __syncthreads();
#pragma unroll
        for (int jt = 0; jt < 8; ++jt) {
            _Float16* cur = ldsB[jt & 1];
            _Float16* nxt = ldsB[(jt + 1) & 1];
            BStage st2;
            if (jt < 7) tileFetch(Wg1t + (jt + 1) * 16 * DDIM, nxt, tid, st2);
            v8f c = {};
#pragma unroll
            for (int t = 0; t < 8; ++t) c = wmma16(xf[t], loadBfragLDS(cur, lr, lhi, t), c);
            if (jt < 7) tileCommit(nxt, tid, st2);
            const float bb = bg1[jt * 16 + lr];
#pragma unroll
            for (int r = 0; r < 8; ++r) {
                float hv = c[r] + bb;
                hv = hv > 0.0f ? hv : 0.0f;
                hw[(r + 8 * lhi) * LDSPITCH + jt * 16 + lr] = (_Float16)hv;
            }
            if (jt < 7) asyncFence();
            __syncthreads();
        }
    }

    // -------- gate GEMM2 (K=128, one 16x16 tile; cols 8..15 padded) + softmax --------
    {
        v8f c = {};
#pragma unroll
        for (int t = 0; t < 4; ++t)
            c = wmma16(loadAfragLDS(hw, lr, lhi, t), loadBfragG(Wg2t, HHID, lr, lhi, t), c);
#pragma unroll
        for (int r = 0; r < 8; ++r) {
            float v = (lr < 8) ? (c[r] + bg2[lr]) : -1e30f;
            float m = v;
            m = fmaxf(m, __shfl_xor(m, 1));
            m = fmaxf(m, __shfl_xor(m, 2));
            m = fmaxf(m, __shfl_xor(m, 4));
            float p = __expf(v - m);
            float s = p;
            s += __shfl_xor(s, 1);
            s += __shfl_xor(s, 2);
            s += __shfl_xor(s, 4);
            float g = p / s;
            if (lr < 8) ldsG[w * 16 + r + 8 * lhi][lr] = g;
        }
    }

    // -------- experts --------
    v8f acc[16];
    {
        v8f z = {};
#pragma unroll
        for (int jt = 0; jt < 16; ++jt) acc[jt] = z;
    }

    for (int e = 0; e < EEXP; ++e) {
        const _Float16* W1e = Wt1 + (size_t)e * DDIM * DDIM;
        const _Float16* W2e = Wt2 + (size_t)e * DDIM * DDIM;
        const float* b1e = b1 + e * DDIM;
        const float* b2e = b2 + e * DDIM;

        // ---- h = relu(x @ W1e + b1e), 16 x 256 -> per-wave LDS tile ----
        {
            BStage st;
            tileFetch(W1e, ldsB[0], tid, st);
            tileCommit(ldsB[0], tid, st);
            asyncFence();
            __syncthreads();
#pragma unroll
            for (int jt = 0; jt < 16; ++jt) {
                _Float16* cur = ldsB[jt & 1];
                _Float16* nxt = ldsB[(jt + 1) & 1];
                BStage st2;
                if (jt < 15) tileFetch(W1e + (jt + 1) * 16 * DDIM, nxt, tid, st2);
                v8f c = {};
#pragma unroll
                for (int t = 0; t < 8; ++t) c = wmma16(xf[t], loadBfragLDS(cur, lr, lhi, t), c);
                if (jt < 15) tileCommit(nxt, tid, st2);
                const float bb = b1e[jt * 16 + lr];
#pragma unroll
                for (int r = 0; r < 8; ++r) {
                    float hv = c[r] + bb;
                    hv = hv > 0.0f ? hv : 0.0f;
                    hw[(r + 8 * lhi) * LDSPITCH + jt * 16 + lr] = (_Float16)hv;
                }
                if (jt < 15) asyncFence();
                __syncthreads();
            }
        }

        // reload h as A fragments (K = 256)
        v16h hf[8];
#pragma unroll
        for (int t = 0; t < 8; ++t) hf[t] = loadAfragLDS(hw, lr, lhi, t);

        float gv[8];
#pragma unroll
        for (int r = 0; r < 8; ++r) gv[r] = ldsG[w * 16 + r + 8 * lhi][e];

        // ---- y = h @ W2e + b2e; out += g[n,e] * y ----
        {
            BStage st;
            tileFetch(W2e, ldsB[0], tid, st);
            tileCommit(ldsB[0], tid, st);
            asyncFence();
            __syncthreads();
#pragma unroll
            for (int jt = 0; jt < 16; ++jt) {
                _Float16* cur = ldsB[jt & 1];
                _Float16* nxt = ldsB[(jt + 1) & 1];
                BStage st2;
                if (jt < 15) tileFetch(W2e + (jt + 1) * 16 * DDIM, nxt, tid, st2);
                v8f c = {};
#pragma unroll
                for (int t = 0; t < 8; ++t) c = wmma16(hf[t], loadBfragLDS(cur, lr, lhi, t), c);
                if (jt < 15) tileCommit(nxt, tid, st2);
                const float bb = b2e[jt * 16 + lr];
#pragma unroll
                for (int r = 0; r < 8; ++r) acc[jt][r] += gv[r] * (c[r] + bb);
                if (jt < 15) asyncFence();
                __syncthreads();
            }
        }
    }

    // -------- store --------
#pragma unroll
    for (int jt = 0; jt < 16; ++jt) {
#pragma unroll
        for (int r = 0; r < 8; ++r)
            out[(size_t)(rowBase + r + 8 * lhi) * DDIM + jt * 16 + lr] = acc[jt][r];
    }
}

extern "C" void kernel_launch(void* const* d_in, const int* in_sizes, int n_in,
                              void* d_out, int out_size, void* d_ws, size_t ws_size,
                              hipStream_t stream) {
    (void)in_sizes; (void)n_in; (void)out_size; (void)ws_size;
    const float* x   = (const float*)d_in[0];
    const float* Wg1 = (const float*)d_in[1];
    const float* bg1 = (const float*)d_in[2];
    const float* Wg2 = (const float*)d_in[3];
    const float* bg2 = (const float*)d_in[4];
    const float* W1  = (const float*)d_in[5];
    const float* b1  = (const float*)d_in[6];
    const float* W2  = (const float*)d_in[7];
    const float* b2  = (const float*)d_in[8];
    float* out = (float*)d_out;

    // workspace carve-up (f16 transposed weights), ~2.2 MB total
    const size_t szW = (size_t)EEXP * DDIM * DDIM;   // 524288 elems
    _Float16* wt1  = (_Float16*)d_ws;
    _Float16* wt2  = wt1 + szW;
    _Float16* wg1t = wt2 + szW;                      // 128*256
    _Float16* wg2t = wg1t + (size_t)HHID * DDIM;     // 16*128

    {
        int total = (int)szW;                        // W1/W2: K=256, N=256, batch=8
        moe_xpose_cvt<<<(total + 255) / 256, 256, 0, stream>>>(W1, wt1, DDIM, DDIM, DDIM, total);
        moe_xpose_cvt<<<(total + 255) / 256, 256, 0, stream>>>(W2, wt2, DDIM, DDIM, DDIM, total);
    }
    {
        int total = HHID * DDIM;                     // Wg1: K=256, N=128
        moe_xpose_cvt<<<(total + 255) / 256, 256, 0, stream>>>(Wg1, wg1t, DDIM, HHID, HHID, total);
    }
    {
        int total = 16 * HHID;                       // Wg2: K=128, Nsrc=8 -> Ndst=16 (zero pad)
        moe_xpose_cvt<<<(total + 255) / 256, 256, 0, stream>>>(Wg2, wg2t, HHID, EEXP, 16, total);
    }

    moe_main_kernel<<<NROWS / 64, 128, 0, stream>>>(x, wt1, wt2, wg1t, wg2t,
                                                    bg1, bg2, b1, b2, out);
}
